// KNRM_83305185673660
// MI455X (gfx1250) — matrix operand order repaired and checked by reading, not compiled, not run on hardware
//
#include <hip/hip_runtime.h>
#include <hip/hip_bf16.h>

typedef __attribute__((ext_vector_type(16))) _Float16 v16h;
typedef __attribute__((ext_vector_type(8)))  _Float16 v8h;
typedef __attribute__((ext_vector_type(8)))  float    v8f;

#define BQ 32
#define BD 512
#define E_REAL 300
#define EP 320          // padded E: 10 chunks of K=32
#define NCH (EP / 32)   // 10
#define NK 11

// ---------------------------------------------------------------------------
// Kernel 1: gather embedding rows for all (query,doc) tokens, L2-normalize,
// store as f16 with zero padding to EP=320. One wave (32 lanes) per token row.
// ---------------------------------------------------------------------------
__global__ void __launch_bounds__(256)
knrm_gather_norm(const int* __restrict__ qtok, const int* __restrict__ dtok,
                 const float* __restrict__ emb,
                 _Float16* __restrict__ qn, _Float16* __restrict__ dn,
                 int n_qrows, int n_drows) {
    const int wave = threadIdx.x >> 5;
    const int lane = threadIdx.x & 31;
    const int row  = blockIdx.x * 8 + wave;
    if (row >= n_qrows + n_drows) return;

    int tok;
    _Float16* outp;
    if (row < n_qrows) {
        tok  = qtok[row];
        outp = qn + (size_t)row * EP;
    } else {
        const int j = row - n_qrows;
        tok  = dtok[j];
        outp = dn + (size_t)j * EP;
    }

    const float* e = emb + (size_t)tok * E_REAL;
    float vals[10];
    float ss = 0.f;
#pragma unroll
    for (int i = 0; i < 10; ++i) {
        const int idx = lane + i * 32;          // covers 0..319
        const float v = (idx < E_REAL) ? e[idx] : 0.f;
        vals[i] = v;
        ss += v * v;
    }
    // wave32 reduction
#pragma unroll
    for (int off = 16; off > 0; off >>= 1)
        ss += __shfl_xor(ss, off, 32);

    const float inv = 1.f / (sqrtf(ss) + 1e-13f);
#pragma unroll
    for (int i = 0; i < 10; ++i) {
        const int idx = lane + i * 32;
        outp[idx] = (_Float16)(vals[i] * inv);
    }
}

// ---------------------------------------------------------------------------
// Kernel 2: per batch b, compute mm = qn @ dn^T (32x512, E=320 reduction) via
// v_wmma_f32_16x16x32_f16, fuse masked RBF kernel pooling over d in registers,
// reduce into LDS, then compute the final score for the batch.
//
// Software pipelined: B fragments are double-buffered across K-chunks (pre-WMMA
// wait covers only the older in-flight pair), and the last prefetch slot of
// each d-tile fetches chunk 0 of the NEXT tile so its latency hides under the
// exp-heavy pooling section.
//
// Fragment placement (wave32; A/B use this code's contiguous k order):
//   A (16x32 f16): lane l -> M = l%16, halves 0..15 = K (l/16)*16+0..15
//   B (32x16 f16): lane l -> N = l%16, halves 0..15 = K (l/16)*16+0..15
//   C (16x16 f32): lane l -> N = l%16, vgpr r -> M = r + 8*(l/16)
// ---------------------------------------------------------------------------
__global__ void __launch_bounds__(256)
knrm_wmma_pool(const int* __restrict__ qtok, const int* __restrict__ dtok,
               const _Float16* __restrict__ qn, const _Float16* __restrict__ dn,
               const float* __restrict__ fcw, const float* __restrict__ fcb,
               float* __restrict__ out) {
    __shared__ float qk[BQ][NK];

    const int b   = blockIdx.x;
    const int tid = threadIdx.x;

    for (int i = tid; i < BQ * NK; i += 256) (&qk[0][0])[i] = 0.f;
    __syncthreads();

    const int wave  = tid >> 5;
    const int lane  = tid & 31;
    const int qtile = wave & 1;       // 0..1  -> q rows [qtile*16, qtile*16+16)
    const int dgrp  = wave >> 1;      // 0..3  -> d tiles [dgrp*8, dgrp*8+8)
    const int lhi   = lane >> 4;      // 0/1
    const int llo   = lane & 15;

    // RBF constants (mus / -1/(2 sigma^2)), n=11 kernels
    constexpr float MU[NK] = {1.0f, 0.9f, 0.7f, 0.5f, 0.3f, 0.1f,
                              -0.1f, -0.3f, -0.5f, -0.7f, -0.9f};
    constexpr float NI[NK] = {-5.0e7f, -50.f, -50.f, -50.f, -50.f, -50.f,
                              -50.f, -50.f, -50.f, -50.f, -50.f};

    // query mask per C row r (q = qtile*16 + lhi*8 + r)
    float qmask[8];
#pragma unroll
    for (int r = 0; r < 8; ++r) {
        const int q = qtile * 16 + lhi * 8 + r;
        qmask[r] = (qtok[b * BQ + q] > 0) ? 1.f : 0.f;
    }

    // doc masks for all 8 tiles this wave owns (d = dgrp*128 + t*16 + llo)
    float dmask[8];
#pragma unroll
    for (int t = 0; t < 8; ++t)
        dmask[t] = (dtok[b * BD + dgrp * 128 + t * 16 + llo] > 0) ? 1.f : 0.f;

    // A fragments: resident in VGPRs for the whole kernel (10 x v16h = 80 VGPRs)
    const _Float16* qbase =
        qn + ((size_t)b * BQ + qtile * 16 + llo) * EP + lhi * 16;
    v16h afrag[NCH];
#pragma unroll
    for (int ch = 0; ch < NCH; ++ch) {
        const v8h alo = *(const v8h*)(qbase + ch * 32);
        const v8h ahi = *(const v8h*)(qbase + ch * 32 + 8);
        afrag[ch] = __builtin_shufflevector(
            alo, ahi, 0, 1, 2, 3, 4, 5, 6, 7, 8, 9, 10, 11, 12, 13, 14, 15);
    }

    float ksum[8][NK];
#pragma unroll
    for (int r = 0; r < 8; ++r)
#pragma unroll
        for (int k = 0; k < NK; ++k) ksum[r][k] = 0.f;

    // per-lane base into B fragments; tile t lives at p0 + t*16*EP
    const _Float16* p0 =
        dn + ((size_t)b * BD + dgrp * 128 + llo) * EP + lhi * 16;

    v16h bcur = *(const v16h*)p0;   // prime the pipeline: tile 0, chunk 0

    for (int t = 0; t < 8; ++t) {
        const _Float16* dcur = p0 + (size_t)t * 16 * EP;
        const _Float16* dnxt = p0 + (size_t)((t + 1) & 7) * 16 * EP;

        v8f c = {};
#pragma unroll
        for (int ch = 0; ch < NCH; ++ch) {
            // prefetch: next chunk of this tile, or chunk 0 of next tile
            const _Float16* np =
                (ch < NCH - 1) ? (dcur + (ch + 1) * 32) : dnxt;
            const v16h bnxt = *(const v16h*)np;
            c = __builtin_amdgcn_wmma_f32_16x16x32_f16(
                false, afrag[ch], false, bcur, (short)0, c, false, false);
            bcur = bnxt;
        }

        const float dm = dmask[t];
#pragma unroll
        for (int r = 0; r < 8; ++r) {
            const float m = c[r];
            const float w = dm * qmask[r];
#pragma unroll
            for (int k = 0; k < NK; ++k) {
                const float diff = m - MU[k];
                ksum[r][k] += w * __expf(diff * diff * NI[k]);
            }
        }
    }

    // accumulate per-q kernel sums into LDS (ds_add_f32 atomics)
#pragma unroll
    for (int r = 0; r < 8; ++r) {
        const int q = qtile * 16 + lhi * 8 + r;
#pragma unroll
        for (int k = 0; k < NK; ++k)
            atomicAdd(&qk[q][k], ksum[r][k]);
    }
    __syncthreads();

    // epilogue: score[b] = sum_q sum_k fcw[k] * 0.01*log(max(qk,1e-10)) + fcb
    if (wave == 0) {
        float partial = 0.f;
#pragma unroll
        for (int k = 0; k < NK; ++k) {
            const float v = qk[lane][k];
            partial += fcw[k] * (0.01f * __logf(fmaxf(v, 1e-10f)));
        }
#pragma unroll
        for (int off = 16; off > 0; off >>= 1)
            partial += __shfl_xor(partial, off, 32);
        if (lane == 0) out[b] = partial + fcb[0];
    }
}

// ---------------------------------------------------------------------------
extern "C" void kernel_launch(void* const* d_in, const int* in_sizes, int n_in,
                              void* d_out, int out_size, void* d_ws, size_t ws_size,
                              hipStream_t stream) {
    const int*   qtok = (const int*)d_in[0];    // [B,32]
    const int*   dtok = (const int*)d_in[1];    // [B,512]
    const float* emb  = (const float*)d_in[2];  // [V,300]
    const float* fcw  = (const float*)d_in[3];  // [1,11]
    const float* fcb  = (const float*)d_in[4];  // [1]
    float*       out  = (float*)d_out;          // [B,1]

    const int n_qrows = in_sizes[0];            // B*32
    const int n_drows = in_sizes[1];            // B*512
    const int B       = n_qrows / BQ;

    // workspace: qn [n_qrows, 320] f16, then dn [n_drows, 320] f16 (~85 MB)
    _Float16* qn = (_Float16*)d_ws;
    _Float16* dn = qn + (size_t)n_qrows * EP;

    const int total_rows = n_qrows + n_drows;
    const int blocksA    = (total_rows + 7) / 8;
    knrm_gather_norm<<<blocksA, 256, 0, stream>>>(qtok, dtok, emb, qn, dn,
                                                  n_qrows, n_drows);

    knrm_wmma_pool<<<B, 256, 0, stream>>>(qtok, dtok, qn, dn, fcw, fcb, out);
}
